// ChaoticLogisticNet_65231963292242
// MI455X (gfx1250) — matrix operand order repaired and checked by reading, not compiled, not run on hardware
//
#include <hip/hip_runtime.h>
#include <hip/hip_bf16.h>

typedef __attribute__((ext_vector_type(16))) _Float16 v16h;
typedef __attribute__((ext_vector_type(8)))  _Float16 v8h;
typedef __attribute__((ext_vector_type(8)))  float    v8f;

#define BATCH    16384
#define WINDOW   512
#define HIDDEN   1024
#define ROWS     16          // batch rows per block (= WMMA M)
#define NTHREADS 1024        // 32 waves (wave32)
#define CPT      16          // cells per thread = ROWS*HIDDEN/NTHREADS
#define PITCH    1032        // f16 row pitch for h tile (bank-conflict padding)

#define R_MIN_C  2.6f
#define DR_C     0.6f        // R_MAX - R_MIN
#define BETA_C   0.1f
#define OMB_C    0.9f        // 1 - BETA
#define EPS_C    1e-6f
#define OME_C    (1.0f - 1e-6f)

// ---- sigmoid -> beta*r_t with all affine constants folded ----------------
// tanh path:  beta*r_t = 0.03*tanh(z/2) + 0.29    (weights pre-scaled by 0.5)
// exp2 path:  s = rcp(1+exp2(-z*log2e)); beta*r_t = 0.06*s + 0.26
#if __has_builtin(__builtin_amdgcn_tanhf)
  #pragma message("sig path: __builtin_amdgcn_tanhf (v_tanh_f32)")
  #define SIG_SCALE 0.5f
  __device__ __forceinline__ float beta_rt(float zp) {
    // BETA*DR*0.5 = 0.03 ; BETA*(R_MIN + DR*0.5) = 0.29
    return fmaf(0.03f, __builtin_amdgcn_tanhf(zp), 0.29f);
  }
#elif __has_builtin(__builtin_amdgcn_tanh_f32)
  #pragma message("sig path: __builtin_amdgcn_tanh_f32 (v_tanh_f32)")
  #define SIG_SCALE 0.5f
  __device__ __forceinline__ float beta_rt(float zp) {
    return fmaf(0.03f, __builtin_amdgcn_tanh_f32(zp), 0.29f);
  }
#elif __has_builtin(__builtin_amdgcn_exp2f) && __has_builtin(__builtin_amdgcn_rcpf)
  #pragma message("sig path: exp2/rcp fallback")
  #define SIG_SCALE (-1.44269504088896340736f)
  __device__ __forceinline__ float beta_rt(float zp) {
    const float s = __builtin_amdgcn_rcpf(1.0f + __builtin_amdgcn_exp2f(zp));
    // BETA*DR = 0.06 ; BETA*R_MIN = 0.26
    return fmaf(0.06f, s, 0.26f);
  }
#else
  #pragma message("sig path: libm fallback")
  #define SIG_SCALE (-1.0f)
  __device__ __forceinline__ float beta_rt(float zp) {
    const float s = 1.0f / (1.0f + __expf(zp));
    return fmaf(0.06f, s, 0.26f);
  }
#endif

__device__ __forceinline__ float clampEps(float v) {
#if __has_builtin(__builtin_amdgcn_fmed3f)
  return __builtin_amdgcn_fmed3f(v, EPS_C, OME_C);   // v_med3_f32 = clamp
#else
  return fminf(fmaxf(v, EPS_C), OME_C);
#endif
}

__global__ __launch_bounds__(NTHREADS, 1)
void chaotic_logistic_scan_kernel(const float* __restrict__ x,
                                  const float* __restrict__ rW,
                                  const float* __restrict__ rb,
                                  const float* __restrict__ oW,
                                  const float* __restrict__ ob,
                                  float* __restrict__ out)
{
    // x tile (16x512 f32 = 32768B) aliased with final h tile (16x1032 f16 = 33024B)
    __shared__ __align__(16) unsigned char s_buf[ROWS * PITCH * 2];
    __shared__ __align__(32) _Float16 s_wout[HIDDEN];
    __shared__ float s_part[32 * ROWS];

    float*    s_x = (float*)s_buf;
    _Float16* s_h = (_Float16*)s_buf;

    const int tid = threadIdx.x;
    const int b0  = blockIdx.x * ROWS;
    const int r   = tid >> 6;     // batch row within tile (64 threads per row)
    const int c   = tid & 63;     // hidden lane; thread owns j = c + 64*k

    // --- stage x tile: rows b0..b0+15 are contiguous in global memory ---
    {
        const float4* xg = (const float4*)(x + (size_t)b0 * WINDOW);
        float4*       xs = (float4*)s_x;
        xs[tid]            = xg[tid];             // 2048 float4 total
        xs[tid + NTHREADS] = xg[tid + NTHREADS];
    }
    // out_W to LDS as f16 (for the WMMA B operand)
    s_wout[tid] = (_Float16)oW[tid];

    // --- per-thread weights in registers, pre-scaled for the sigmoid path ---
    float wk[CPT], bk[CPT], h[CPT];
    #pragma unroll
    for (int k = 0; k < CPT; ++k) {
        const int j = c + 64 * k;
        wk[k] = SIG_SCALE * rW[j];
        bk[k] = SIG_SCALE * rb[j];
        h[k]  = 0.5f;
    }
    __syncthreads();

    // --- the scan: 512 steps, 16 independent cells per thread (ILP=16) ---
    // per cell-step: fma, tanh, fma, fma, mul, fma, med3  -> 7 VALU (1 TRANS)
    const float* xrow = s_x + r * WINDOW;
    for (int t = 0; t < WINDOW; t += 4) {
        const float4 u4 = *(const float4*)(xrow + t);   // 1 LDS read / 64 cell-steps
        const float us[4] = {u4.x, u4.y, u4.z, u4.w};
        #pragma unroll
        for (int tt = 0; tt < 4; ++tt) {
            const float u = us[tt];
            #pragma unroll
            for (int k = 0; k < CPT; ++k) {
                const float rtb = beta_rt(fmaf(u, wk[k], bk[k]));  // beta*r_t
                const float hv  = h[k];
                const float p   = fmaf(-hv, hv, hv);               // h*(1-h)
                const float nh  = fmaf(rtb, p, OMB_C * hv);        // 0.9h + beta*r_t*h(1-h)
                h[k] = clampEps(nh);
            }
        }
    }

    // --- dump final h to LDS as f16 (reuses the x region) ---
    __syncthreads();
    #pragma unroll
    for (int k = 0; k < CPT; ++k)
        s_h[r * PITCH + c + 64 * k] = (_Float16)h[k];
    __syncthreads();

    // --- WMMA epilogue: out[m] = sum_n h[m,n]*wout[n]; one 16x16x32 per wave ---
    const int  wave = tid >> 5;
    const int  lane = tid & 31;
    const int  m    = lane & 15;
    const bool up   = lane >= 16;
    const int  hi   = up ? 8 : 0;    // A-matrix K striping (16-bit A 16x32 layout)
    const int  kb   = up ? 16 : 0;   // unused: B is loaded with hi, same K order as A
    const int  c32  = wave * 32;     // this wave's 32-wide hidden chunk

    // A lane data: halves h[m][c32+hi+0..7] then h[m][c32+hi+16..23]
    const _Float16* hrow = s_h + m * PITCH + c32 + hi;
    const v8h alo = *(const v8h*)(hrow);
    const v8h ahi = *(const v8h*)(hrow + 16);
    v16h a;
    #pragma unroll
    for (int e = 0; e < 8; ++e) { a[e] = alo[e]; a[8 + e] = ahi[e]; }
    // B[k,n] = wout[c32+k] broadcast over n -> halves wout[c32+hi+0..7] then wout[c32+hi+16..23]
    const v8h blo = *(const v8h*)(s_wout + c32 + hi);
    const v8h bhi = *(const v8h*)(s_wout + c32 + hi + 16);
    v16h b;
    #pragma unroll
    for (int e = 0; e < 8; ++e) { b[e] = blo[e]; b[8 + e] = bhi[e]; }

    v8f acc = {};
    acc = __builtin_amdgcn_wmma_f32_16x16x32_f16(
              /*neg_a=*/false, a, /*neg_b=*/false, b,
              /*c_mod=*/(short)0, acc, /*reuse_a=*/false, /*reuse_b=*/false);

    // D columns are identical; lane0 holds M=0..7, lane16 holds M=8..15
    if (lane == 0) {
        #pragma unroll
        for (int v = 0; v < 8; ++v) s_part[wave * ROWS + v] = acc[v];
    } else if (lane == 16) {
        #pragma unroll
        for (int v = 0; v < 8; ++v) s_part[wave * ROWS + 8 + v] = acc[v];
    }
    __syncthreads();

    // cross-wave reduce: 32 partials per batch row
    if (tid < ROWS) {
        float sum = 0.0f;
        #pragma unroll
        for (int w2 = 0; w2 < 32; ++w2) sum += s_part[w2 * ROWS + tid];
        out[b0 + tid] = sum + ob[0];
    }
}

extern "C" void kernel_launch(void* const* d_in, const int* in_sizes, int n_in,
                              void* d_out, int out_size, void* d_ws, size_t ws_size,
                              hipStream_t stream) {
    (void)in_sizes; (void)n_in; (void)out_size; (void)d_ws; (void)ws_size;
    const float* x  = (const float*)d_in[0];
    const float* rW = (const float*)d_in[1];
    const float* rb = (const float*)d_in[2];
    const float* oW = (const float*)d_in[3];
    const float* ob = (const float*)d_in[4];
    float* out = (float*)d_out;

    dim3 grid(BATCH / ROWS);   // 1024 blocks
    dim3 block(NTHREADS);      // 1024 threads = 32 wave32
    hipLaunchKernelGGL(chaotic_logistic_scan_kernel, grid, block, 0, stream,
                       x, rW, rb, oW, ob, out);
}
